// SmoothRankAP_88313117541018
// MI455X (gfx1250) — compile-verified, matrix operand order checked by reading, not run on hardware
//
#include <hip/hip_runtime.h>

typedef __attribute__((ext_vector_type(16))) _Float16     v16h;
typedef __attribute__((ext_vector_type(8)))  float        v8f;
typedef __attribute__((ext_vector_type(4)))  float        v4f;
typedef __attribute__((ext_vector_type(8)))  unsigned int v8u;
typedef __attribute__((ext_vector_type(4)))  unsigned int v4u;

#define BSZ      512
#define HALF_IT  50.0f       // 0.5 / tau  (tanh form of sigmoid, tau = 0.01)

// ---------------------------------------------------------------------------
// Prep: target f32 -> per-f16-slot bitmasks (0xFFFF / 0x0000) + row sums.
// masks[r*512+c] selects sig bits for the positive sum; sum_t[r] = sum_c t[r,c].
// ---------------------------------------------------------------------------
__global__ __launch_bounds__(256) void smoothap_prep(const float* __restrict__ target,
                                                     unsigned int* __restrict__ masks32,
                                                     float* __restrict__ sum_t) {
    __shared__ float red[256];
    const int r   = blockIdx.x;
    const int tid = threadIdx.x;
    const float2 t2 = ((const float2*)(target + (size_t)r * BSZ))[tid];
    unsigned int m = (t2.x != 0.0f ? 0x0000FFFFu : 0u) |
                     (t2.y != 0.0f ? 0xFFFF0000u : 0u);
    masks32[(size_t)r * (BSZ / 2) + tid] = m;
    red[tid] = t2.x + t2.y;
    __syncthreads();
#pragma unroll
    for (int s = 128; s > 0; s >>= 1) {
        if (tid < s) red[tid] += red[tid + s];
        __syncthreads();
    }
    if (tid == 0) sum_t[r] = red[0];
}

// ---------------------------------------------------------------------------
// Main: one block per query i; 8 waves x 4 j-tiles of 16 rows.
// Inner loop: x = 50*(s_k - s_j); th = v_tanh_f32(x); A_all = f16(th);
// A_pos = A_all & mask. WMMA against B = all-0.5 accumulates
// c_all = sum_k 0.5*th and c_pos = sum_k 0.5*th*t; the affine sigmoid shift
// is restored in the epilogue: sum sig = c_all + 256, sum sig*t = c_pos + 0.5*sum_t[j].
// ---------------------------------------------------------------------------
__global__ __launch_bounds__(256) void smoothap_main(const float* __restrict__ scores,
                                                     const float* __restrict__ target,
                                                     const unsigned short* __restrict__ masks,
                                                     const float* __restrict__ sum_t,
                                                     float* __restrict__ ap_out) {
    __shared__ float srow50[BSZ];     // 50 * scores[i, :]
    __shared__ float red_ap[256];

    const int i   = blockIdx.x;
    const int tid = threadIdx.x;

    {   // stage + pre-scale scores row i
        const float2 s2 = ((const float2*)(scores + (size_t)i * BSZ))[tid];
        ((float2*)srow50)[tid] = make_float2(s2.x * HALF_IT, s2.y * HALF_IT);
    }
    __syncthreads();

    const int lane   = tid & 31;
    const int wave   = tid >> 5;
    const int laneLo = lane & 15;
    const int laneHi = lane >> 4;

    float ap_acc = 0.0f;

    v16h bhalf;
#pragma unroll
    for (int e = 0; e < 16; ++e) bhalf[e] = (_Float16)0.5f;

    for (int jt = 0; jt < 4; ++jt) {
        const int tileBase = (wave * 4 + jt) * 16;
        const int j        = tileBase + laneLo;       // this lane's A-matrix row
        const float sj50   = srow50[j];
        const unsigned short* mrow = masks + (size_t)j * BSZ;

        v8f c_all = {};   // sum_k 0.5*th
        v8f c_pos = {};   // sum_k 0.5*th*t

        for (int kb = 0; kb < BSZ; kb += 32) {
            const int ofs = kb + laneHi * 8;
            // f16 A 16x32 layout: element e -> k = kb + e + (e>=8?8:0) + laneHi*8
            v4f s0 = *(const v4f*)(srow50 + ofs);
            v4f s1 = *(const v4f*)(srow50 + ofs + 4);
            v4f s2 = *(const v4f*)(srow50 + ofs + 16);
            v4f s3 = *(const v4f*)(srow50 + ofs + 20);
            v4u m0 = *(const v4u*)(mrow + ofs);        // masks for k = ofs..ofs+7
            v4u m1 = *(const v4u*)(mrow + ofs + 16);   // masks for k = ofs+16..ofs+23

            v16h a_all;
#pragma unroll
            for (int e = 0; e < 16; ++e) {
                float sk50 = (e < 4) ? s0[e & 3] : (e < 8) ? s1[e & 3]
                           : (e < 12) ? s2[e & 3] : s3[e & 3];
                // th = tanh((s_k - s_j) * 50)  (v_tanh_f32, gfx1250)
                a_all[e] = (_Float16)__builtin_amdgcn_tanhf(sk50 - sj50);
            }
            v8u au = __builtin_bit_cast(v8u, a_all);
            v8u mu;
            mu[0] = m0[0]; mu[1] = m0[1]; mu[2] = m0[2]; mu[3] = m0[3];
            mu[4] = m1[0]; mu[5] = m1[1]; mu[6] = m1[2]; mu[7] = m1[3];
            v16h a_pos = __builtin_bit_cast(v16h, au & mu);   // th * t (bitselect)

            // k-reduction on the matrix unit: D = A x 0.5 + C
            c_all = __builtin_amdgcn_wmma_f32_16x16x32_f16(false, a_all, false, bhalf,
                                                           (short)0, c_all, false, false);
            c_pos = __builtin_amdgcn_wmma_f32_16x16x32_f16(false, a_pos, false, bhalf,
                                                           (short)0, c_pos, false, false);
        }

        // All 16 N-columns of D are identical; lane 0 holds tile rows 0..7 in
        // c[0..7], lane 16 holds rows 8..15.
        if (laneLo == 0) {
#pragma unroll
            for (int r = 0; r < 8; ++r) {
                const int   jrow  = tileBase + laneHi * 8 + r;
                const float t     = target[(size_t)i * BSZ + jrow];
                const float stj   = sum_t[jrow];
                // sum_k sigmoid      = c_all + 256   ; sim_all_rk = that + 0.5
                // sum_k sigmoid*t    = c_pos + 0.5*stj
                // sim_pos_rk         = (sum_pos - 0.5 + t) * t   (t factor => 0 when t==0)
                const float denom = c_all[r] + 256.5f;
                const float num   = (c_pos[r] + 0.5f * stj - 0.5f + t) * t;
                ap_acc += num * __builtin_amdgcn_rcpf(denom);
            }
        }
    }

    red_ap[tid] = ap_acc;
    __syncthreads();
#pragma unroll
    for (int s = 128; s > 0; s >>= 1) {
        if (tid < s) red_ap[tid] += red_ap[tid + s];
        __syncthreads();
    }
    if (tid == 0) ap_out[i] = red_ap[0] * __builtin_amdgcn_rcpf(sum_t[i]);
}

__global__ __launch_bounds__(256) void smoothap_reduce(const float* __restrict__ ap,
                                                       float* __restrict__ out) {
    __shared__ float red[256];
    const int tid = threadIdx.x;
    red[tid] = ap[tid] + ap[tid + 256];
    __syncthreads();
#pragma unroll
    for (int s = 128; s > 0; s >>= 1) {
        if (tid < s) red[tid] += red[tid + s];
        __syncthreads();
    }
    if (tid == 0) out[0] = 1.0f - red[0] * (1.0f / (float)BSZ);
}

extern "C" void kernel_launch(void* const* d_in, const int* in_sizes, int n_in,
                              void* d_out, int out_size, void* d_ws, size_t ws_size,
                              hipStream_t stream) {
    const float* scores = (const float*)d_in[0];   // [512,512] f32
    const float* target = (const float*)d_in[1];   // [512,512] f32

    // Workspace layout (recomputed every call; no cross-call state):
    //   [0, 512KB)            u16 masks  (512*512)
    //   [512KB, 512KB+2KB)    f32 sum_t  (512)
    //   [514KB, 514KB+2KB)    f32 ap     (512)
    char* ws = (char*)d_ws;
    unsigned short* masks = (unsigned short*)ws;
    float* sumt = (float*)(ws + (size_t)BSZ * BSZ * sizeof(unsigned short));
    float* ap   = sumt + BSZ;
    float* out  = (float*)d_out;

    smoothap_prep<<<BSZ, 256, 0, stream>>>(target, (unsigned int*)masks, sumt);
    smoothap_main<<<BSZ, 256, 0, stream>>>(scores, target, masks, sumt, ap);
    smoothap_reduce<<<1, 256, 0, stream>>>(ap, out);
}
